// sLSTMCell_27573690040413
// MI455X (gfx1250) — matrix operand order repaired and checked by reading, not compiled, not run on hardware
//
#include <hip/hip_runtime.h>
#include <hip/hip_bf16.h>
#include <math.h>

// sLSTM cell: B=4096, D_IN=D_H=2048, fp32 in/out.
// pre_g = x@W_g + b_g + h_prev@U_g  for g in {i,f,o,z}, then pointwise.
// Fused 4-gate bf16 WMMA GEMM (fp32 accumulate) + fused epilogue.

#define B_ROWS 4096
#define D      2048
#define BM     128   // workgroup M tile (batch rows)
#define BN     64    // workgroup N tile (hidden cols) per gate
#define KC     32    // K step (matches wmma 16x16x32 bf16)
#define LDT    40    // padded LDS row stride in bf16 (80B = 20 dwords, conflict-free)
#define NG     4     // gates
#define NK     (2 * (D / KC))   // 128 K-steps (x phase + h phase)

typedef __attribute__((ext_vector_type(16))) __bf16 v16bf;
typedef __attribute__((ext_vector_type(8)))  __bf16 bf16x8;
typedef __attribute__((ext_vector_type(4)))  __bf16 bf16x4;
typedef __attribute__((ext_vector_type(8)))  float  v8f;
typedef __attribute__((ext_vector_type(4)))  float  f32x4;

union FragU { v16bf v; bf16x8 h[2]; };

struct SLSTMParams {
  const float* x;
  const float* hp;
  const float* cp;
  const float* np;
  const float* W[NG];
  const float* bias[NG];
  const float* U[NG];
  float* out;
};

__global__ __launch_bounds__(256, 1)
void slstm_wmma_kernel(SLSTMParams p)
{
  // Double-buffered LDS tiles. A: [BM x KC] row-major (rows = batch).
  // B: [NG*BN x KC] stored TRANSPOSED (rows = output column N), so both A and B
  // fragments load as two 16B ds reads per lane matching the WMMA VGPR layout.
  __shared__ __align__(16) __bf16 Asm[2][BM * LDT];
  __shared__ __align__(16) __bf16 Bsm[2][NG * BN * LDT];

  const int tid   = threadIdx.x;
  const int lane  = tid & 31;
  const int l15   = lane & 15;
  const int khalf = lane >> 4;          // 0: lanes 0-15, 1: lanes 16-31
  const int wave  = tid >> 5;
  const int waveM = wave & 3;           // 4 wave-rows * 32 rows = 128
  const int waveN = wave >> 2;          // 2 wave-cols * 32 cols = 64

  const int mBase = blockIdx.y * BM;
  const int nBase = blockIdx.x * BN;

  v8f acc[NG][2][2];
  #pragma unroll
  for (int g = 0; g < NG; ++g)
    #pragma unroll
    for (int ms = 0; ms < 2; ++ms)
      #pragma unroll
      for (int ns = 0; ns < 2; ++ns)
        acc[g][ms][ns] = (v8f){0.f,0.f,0.f,0.f,0.f,0.f,0.f,0.f};

  // ---- per-thread constant addressing (computed once) ----
  // A tile: thread owns 4 float4s: rows (tid>>3)+i*32, cols (tid&7)*4 .. +3
  const int arow  = tid >> 3;
  const int acol  = (tid & 7) * 4;
  const unsigned aoff0 = (unsigned)(mBase + arow) * D + (unsigned)acol;

  // B tiles: thread owns one 4K x 4N block in each of two gate streams.
  // stream j=0 -> gate g0 in {0,1}, j=1 -> gate g0+2. 16 consecutive threads
  // cover 16 consecutive N-quads of one K row (256B contiguous per load).
  const bool hi = (tid & 128) != 0;
  const int  g0 = hi ? 1 : 0;
  const int  g1 = g0 + 2;
  const float* Wj0 = hi ? p.W[1] : p.W[0];
  const float* Uj0 = hi ? p.U[1] : p.U[0];
  const float* Wj1 = hi ? p.W[3] : p.W[2];
  const float* Uj1 = hi ? p.U[3] : p.U[2];
  const int blk = tid & 127;
  const int kb  = blk >> 4;             // K-row base = kb*4
  const int nb  = blk & 15;             // N base     = nb*4
  const unsigned boff = (unsigned)(kb * 4) * D + (unsigned)(nBase + nb * 4);

  f32x4 aR[4];
  f32x4 bR[2][4];   // [stream][K row within 4K x 4N block]

  auto load_tile = [&](int ks) {
    const int ph = ks >> 6;                 // 64 K-steps per phase
    const int k0 = (ks & 63) * KC;
    const float* Ab = (ph ? p.hp : p.x) + k0;
    #pragma unroll
    for (int i = 0; i < 4; ++i)
      aR[i] = *(const f32x4*)(Ab + aoff0 + (unsigned)(i * 32) * D);
    const float* B0 = (ph ? Uj0 : Wj0) + (size_t)k0 * D + boff;
    const float* B1 = (ph ? Uj1 : Wj1) + (size_t)k0 * D + boff;
    #pragma unroll
    for (int r = 0; r < 4; ++r) {
      bR[0][r] = *(const f32x4*)(B0 + (unsigned)(r * D));
      bR[1][r] = *(const f32x4*)(B1 + (unsigned)(r * D));
    }
  };

  auto lds_store = [&](int buf) {
    #pragma unroll
    for (int i = 0; i < 4; ++i) {
      bf16x4 t;
      t[0] = (__bf16)aR[i][0]; t[1] = (__bf16)aR[i][1];
      t[2] = (__bf16)aR[i][2]; t[3] = (__bf16)aR[i][3];
      *(bf16x4*)(&Asm[buf][(arow + i * 32) * LDT + acol]) = t;
    }
    #pragma unroll
    for (int j = 0; j < 2; ++j) {
      const int grow = (j ? g1 : g0) * BN + nb * 4;
      #pragma unroll
      for (int e = 0; e < 4; ++e) {     // register transpose: 4 K for one N
        bf16x4 t;
        t[0] = (__bf16)bR[j][0][e]; t[1] = (__bf16)bR[j][1][e];
        t[2] = (__bf16)bR[j][2][e]; t[3] = (__bf16)bR[j][3][e];
        *(bf16x4*)(&Bsm[buf][(grow + e) * LDT + kb * 4]) = t;
      }
    }
  };

  load_tile(0);

  for (int ks = 0; ks < NK; ++ks) {
    const int buf = ks & 1;
    lds_store(buf);
    if (ks + 1 < NK) load_tile(ks + 1);   // global prefetch overlaps compute
    __syncthreads();                       // single barrier per K-step

    // A fragments (16x32 bf16): lane<16 -> K 0-7 & 16-23; lane>=16 -> 8-15 & 24-31
    FragU a0, a1;
    {
      const __bf16* b0 = &Asm[buf][(waveM * 32 + l15) * LDT + khalf * 8];
      a0.h[0] = *(const bf16x8*)(b0);
      a0.h[1] = *(const bf16x8*)(b0 + 16);
      const __bf16* b1 = &Asm[buf][(waveM * 32 + 16 + l15) * LDT + khalf * 8];
      a1.h[0] = *(const bf16x8*)(b1);
      a1.h[1] = *(const bf16x8*)(b1 + 16);
    }

    #pragma unroll
    for (int g = 0; g < NG; ++g) {
      #pragma unroll
      for (int ns = 0; ns < 2; ++ns) {
        // B fragment (32x16 bf16, N = lane%16): lane<16 -> K 0-7 & 16-23, lane>=16 -> 8-15 & 24-31
        FragU bf;
        const __bf16* bb =
            &Bsm[buf][(g * BN + waveN * 32 + ns * 16 + l15) * LDT + khalf * 8];
        bf.h[0] = *(const bf16x8*)(bb);
        bf.h[1] = *(const bf16x8*)(bb + 16);
        acc[g][0][ns] = __builtin_amdgcn_wmma_f32_16x16x32_bf16(
            false, a0.v, false, bf.v, (short)0, acc[g][0][ns], false, false);
        acc[g][1][ns] = __builtin_amdgcn_wmma_f32_16x16x32_bf16(
            false, a1.v, false, bf.v, (short)0, acc[g][1][ns], false, false);
      }
    }
  }

  // Fused epilogue. C/D layout: VGPR r, lane l -> M = r + 8*(l>>4), N = l&15.
  float* outH = p.out;
  float* outC = p.out + (size_t)B_ROWS * D;
  float* outN = p.out + (size_t)2 * B_ROWS * D;
  const int rowBase = mBase + waveM * 32 + khalf * 8;

  #pragma unroll
  for (int ns = 0; ns < 2; ++ns) {
    const int col = nBase + waveN * 32 + ns * 16 + l15;
    const float bi  = p.bias[0][col];
    const float bf_ = p.bias[1][col];
    const float bo  = p.bias[2][col];
    const float bz  = p.bias[3][col];
    #pragma unroll
    for (int ms = 0; ms < 2; ++ms) {
      #pragma unroll
      for (int r = 0; r < 8; ++r) {
        const int row = rowBase + ms * 16 + r;
        const size_t idx = (size_t)row * D + col;
        const float gi = __expf(acc[0][ms][ns][r] + bi);
        const float gf = __expf(acc[1][ms][ns][r] + bf_);
        const float go = 1.0f / (1.0f + __expf(-(acc[2][ms][ns][r] + bo)));
        const float gz = tanhf(acc[3][ms][ns][r] + bz);
        const float cprev = p.cp[idx];
        const float nprev = p.np[idx];
        const float cN = gf * cprev + gi * gz;
        const float nN = gf * nprev + gi;
        const float hN = go * (cN / (nN + 1e-6f));
        outH[idx] = hN;
        outC[idx] = cN;
        outN[idx] = nN;
      }
    }
  }
}

extern "C" void kernel_launch(void* const* d_in, const int* in_sizes, int n_in,
                              void* d_out, int out_size, void* d_ws, size_t ws_size,
                              hipStream_t stream) {
  (void)in_sizes; (void)n_in; (void)d_ws; (void)ws_size; (void)out_size;
  SLSTMParams p;
  p.x  = (const float*)d_in[0];
  p.hp = (const float*)d_in[1];
  p.cp = (const float*)d_in[2];
  p.np = (const float*)d_in[3];
  // dict order: W_g, b_g, U_g for g in {i, f, o, z}
  for (int g = 0; g < NG; ++g) {
    p.W[g]    = (const float*)d_in[4 + 3 * g];
    p.bias[g] = (const float*)d_in[5 + 3 * g];
    p.U[g]    = (const float*)d_in[6 + 3 * g];
  }
  p.out = (float*)d_out;

  dim3 grid(D / BN, B_ROWS / BM);   // (32, 32)
  dim3 block(256, 1, 1);
  slstm_wmma_kernel<<<grid, block, 0, stream>>>(p);
}
